// SpectralConv1d_83296595739171
// MI455X (gfx1250) — matrix operand order repaired and checked by reading, not compiled, not run on hardware
//
#include <hip/hip_runtime.h>
#include <hip/hip_bf16.h>

typedef __attribute__((ext_vector_type(16))) _Float16 v16h;
typedef __attribute__((ext_vector_type(8)))  float    v8f;

#define N_TIME   8192
#define BATCH    64
#define IN_CH    64
#define OUT_CH   64
#define MODES    16
#define NFREQ2   (2*MODES)        // 32 interleaved re/im columns
#define M_ROWS   (BATCH*IN_CH)    // 4096
#define RB_COUNT (M_ROWS/16)      // 256 row-blocks of 16
#define KCHUNKS  (N_TIME/32)      // 256 K-chunks for phase A
#define NTILES   (N_TIME/16)      // 512 n-tiles for phase C
#define BAS_DW   (KCHUNKS*2*32*8) // 131072 dwords (512 KB)
#define BINV_DW  (NTILES*32*8)    // 131072 dwords (512 KB)

#define TWO_PI_OVER_N 7.66990393942820614e-4f

static __device__ inline unsigned pk(_Float16 a, _Float16 b) {
    union { _Float16 h[2]; unsigned u; } x;
    x.h[0] = a; x.h[1] = b; return x.u;
}

// ---------------------------------------------------------------------------
// Generate forward DFT basis (pre-swizzled WMMA B layout, K=time) and inverse
// basis (pre-swizzled WMMA B layout, K=freq-col), each split hi/lo f16.
// ---------------------------------------------------------------------------
__global__ __launch_bounds__(256) void gen_bases(unsigned* __restrict__ bas_hi,
                                                 unsigned* __restrict__ bas_lo,
                                                 unsigned* __restrict__ binv_hi,
                                                 unsigned* __restrict__ binv_lo) {
    int t = blockIdx.x * blockDim.x + threadIdx.x;
    float val0, val1;
    unsigned *ph, *pl;
    if (t < BAS_DW) {
        // forward basis: dword d = (((chunk*2+tile)*32)+lane)*8 + v
        int d = t;
        int v = d & 7, lane = (d >> 3) & 31, tile = (d >> 8) & 1, chunk = d >> 9;
        int n0  = chunk * 32 + 2 * v + ((lane & 16) ? 8 : 0) + ((v & 4) ? 8 : 0); // time index (K dim)
        int col = tile * 16 + (lane & 15);                     // freq column (N dim)
        int m = col >> 1;
        int p0 = (m * n0) & (N_TIME - 1);
        int p1 = (m * (n0 + 1)) & (N_TIME - 1);
        float a0 = p0 * TWO_PI_OVER_N, a1 = p1 * TWO_PI_OVER_N;
        if (col & 1) { val0 = -sinf(a0); val1 = -sinf(a1); }
        else         { val0 =  cosf(a0); val1 =  cosf(a1); }
        ph = bas_hi + d; pl = bas_lo + d;
    } else {
        // inverse basis: dword d = ((ntile*32)+lane)*8 + v
        int d = t - BAS_DW;
        int v = d & 7, lane = (d >> 3) & 31, ntile = d >> 8;
        int k0 = 2 * v + ((lane & 16) ? 8 : 0) + ((v & 4) ? 8 : 0); // freq column (K dim)
        int n  = ntile * 16 + (lane & 15);                     // time index (N dim)
        int m0 = k0 >> 1, m1 = (k0 + 1) >> 1;
        int p0 = (m0 * n) & (N_TIME - 1);
        int p1 = (m1 * n) & (N_TIME - 1);
        float a0 = p0 * TWO_PI_OVER_N, a1 = p1 * TWO_PI_OVER_N;
        val0 = (k0 & 1)       ? -sinf(a0) : cosf(a0);
        val1 = ((k0 + 1) & 1) ? -sinf(a1) : cosf(a1);
        ph = binv_hi + d; pl = binv_lo + d;
    }
    _Float16 h0 = (_Float16)val0, h1 = (_Float16)val1;
    _Float16 l0 = (_Float16)(val0 - (float)h0);
    _Float16 l1 = (_Float16)(val1 - (float)h1);
    *ph = pk(h0, h1);
    *pl = pk(l0, l1);
}

// ---------------------------------------------------------------------------
// Phase A: X[(b,i),0:32] = x[(b,i),:] @ basis   (M=4096,K=8192,N=32)
// 8 waves per block split K; split-f16 WMMA, f32 accumulate, LDS reduction.
// ---------------------------------------------------------------------------
__global__ __launch_bounds__(256) void phaseA(const float* __restrict__ x,
                                              const unsigned* __restrict__ bas_hi,
                                              const unsigned* __restrict__ bas_lo,
                                              float* __restrict__ Xw) {
    int rb = blockIdx.x;
    int tid = threadIdx.x;
    int wave = tid >> 5, lane = tid & 31;
    int lr = lane & 15, hs = lane >> 4;
    const float* xrow = x + (size_t)(rb * 16 + lr) * N_TIME;
    int ka0 = hs ? 8 : 0, ka1 = hs ? 24 : 16;   // A-operand K-halves for this lane group

    v8f c0 = {}, c1 = {};
    int c_begin = wave * (KCHUNKS / 8);
    int c_end   = c_begin + (KCHUNKS / 8);
    for (int ch = c_begin; ch < c_end; ++ch) {
        int kb = ch * 32;
        float4 a0 = *(const float4*)(xrow + kb + ka0);
        float4 a1 = *(const float4*)(xrow + kb + ka0 + 4);
        float4 a2 = *(const float4*)(xrow + kb + ka1);
        float4 a3 = *(const float4*)(xrow + kb + ka1 + 4);
        float vals[16] = {a0.x,a0.y,a0.z,a0.w, a1.x,a1.y,a1.z,a1.w,
                          a2.x,a2.y,a2.z,a2.w, a3.x,a3.y,a3.z,a3.w};
        v16h ah, al;
        #pragma unroll
        for (int j = 0; j < 16; ++j) {
            _Float16 h = (_Float16)vals[j];
            ah[j] = h;
            al[j] = (_Float16)(vals[j] - (float)h);
        }
        v16h b0h = *(const v16h*)(bas_hi + (((size_t)ch * 2 + 0) * 32 + lane) * 8);
        v16h b1h = *(const v16h*)(bas_hi + (((size_t)ch * 2 + 1) * 32 + lane) * 8);
        v16h b0l = *(const v16h*)(bas_lo + (((size_t)ch * 2 + 0) * 32 + lane) * 8);
        v16h b1l = *(const v16h*)(bas_lo + (((size_t)ch * 2 + 1) * 32 + lane) * 8);
        c0 = __builtin_amdgcn_wmma_f32_16x16x32_f16(false, ah, false, b0h, (short)0, c0, false, false);
        c0 = __builtin_amdgcn_wmma_f32_16x16x32_f16(false, al, false, b0h, (short)0, c0, false, false);
        c0 = __builtin_amdgcn_wmma_f32_16x16x32_f16(false, ah, false, b0l, (short)0, c0, false, false);
        c1 = __builtin_amdgcn_wmma_f32_16x16x32_f16(false, ah, false, b1h, (short)0, c1, false, false);
        c1 = __builtin_amdgcn_wmma_f32_16x16x32_f16(false, al, false, b1h, (short)0, c1, false, false);
        c1 = __builtin_amdgcn_wmma_f32_16x16x32_f16(false, ah, false, b1l, (short)0, c1, false, false);
    }

    __shared__ float red[16 * NFREQ2];   // 512 floats
    red[tid] = 0.0f;
    red[tid + 256] = 0.0f;
    __syncthreads();
    #pragma unroll
    for (int r = 0; r < 8; ++r) {
        atomicAdd(&red[(r + hs * 8) * NFREQ2 + lr],      c0[r]);
        atomicAdd(&red[(r + hs * 8) * NFREQ2 + 16 + lr], c1[r]);
    }
    __syncthreads();
    Xw[(size_t)rb * 512 + tid]       = red[tid];
    Xw[(size_t)rb * 512 + 256 + tid] = red[256 + tid];
}

// ---------------------------------------------------------------------------
// Phase B: per-mode complex channel mixing; outputs split-f16 in phase-C
// WMMA A-operand swizzle (dword v of lane = {Re,Im} of one mode).
// Folds irfft scale a_m*4096 = {0.5, 1}.
// ---------------------------------------------------------------------------
__global__ __launch_bounds__(1024) void phaseB(const float* __restrict__ Xw,
                                               const float* __restrict__ wr,
                                               const float* __restrict__ wi,
                                               unsigned* __restrict__ c2h,
                                               unsigned* __restrict__ c2l) {
    int b = blockIdx.x;
    int tid = threadIdx.x;          // = o*16 + m
    int o = tid >> 4, m = tid & 15;
    __shared__ float Xs[IN_CH * NFREQ2];   // 2048 floats
    Xs[tid]        = Xw[(size_t)b * 2048 + tid];
    Xs[tid + 1024] = Xw[(size_t)b * 2048 + 1024 + tid];
    __syncthreads();

    float fr = 0.0f, fi = 0.0f;
    #pragma unroll 4
    for (int i = 0; i < IN_CH; ++i) {
        float xr  = Xs[i * NFREQ2 + 2 * m];
        float xi  = Xs[i * NFREQ2 + 2 * m + 1];
        float wre = wr[((size_t)i * OUT_CH + o) * MODES + m];
        float wim = wi[((size_t)i * OUT_CH + o) * MODES + m];
        fr += xr * wre - xi * wim;
        fi += xr * wim + xi * wre;
    }
    float s = (m == 0) ? 0.5f : 1.0f;   // a_m * 4096
    fr *= s; fi *= s;

    _Float16 hr  = (_Float16)fr,  hi2 = (_Float16)fi;
    _Float16 lr2 = (_Float16)(fr - (float)hr);
    _Float16 li2 = (_Float16)(fi - (float)hi2);

    int row = b * OUT_CH + o;
    int rb = row >> 4, M = row & 15;
    int lane = M + ((((m >= 4) && (m < 8)) || (m >= 12)) ? 16 : 0);
    int v = (m < 4) ? m : ((m < 12) ? m - 4 : m - 8);
    size_t idx = ((size_t)rb * 32 + lane) * 8 + v;
    c2h[idx] = pk(hr, hi2);
    c2l[idx] = pk(lr2, li2);
}

// ---------------------------------------------------------------------------
// Phase C: y = C2 @ binv  (M=4096,K=32,N=8192); one split-f16 WMMA triple per
// 16x16 output tile, scaled by 1/4096 on the f32 store.
// ---------------------------------------------------------------------------
__global__ __launch_bounds__(256) void phaseC(const unsigned* __restrict__ c2h,
                                              const unsigned* __restrict__ c2l,
                                              const unsigned* __restrict__ binv_hi,
                                              const unsigned* __restrict__ binv_lo,
                                              float* __restrict__ out) {
    int rb = blockIdx.x;
    int tid = threadIdx.x;
    int wave = tid >> 5, lane = tid & 31;
    int lr = lane & 15, hs = lane >> 4;

    v16h ah = *(const v16h*)(c2h + ((size_t)rb * 32 + lane) * 8);
    v16h al = *(const v16h*)(c2l + ((size_t)rb * 32 + lane) * 8);

    int base_t = (blockIdx.y * 8 + wave) * 4;
    for (int t = 0; t < 4; ++t) {
        int ntile = base_t + t;
        v16h bh = *(const v16h*)(binv_hi + ((size_t)ntile * 32 + lane) * 8);
        v16h bl = *(const v16h*)(binv_lo + ((size_t)ntile * 32 + lane) * 8);
        v8f c = {};
        c = __builtin_amdgcn_wmma_f32_16x16x32_f16(false, ah, false, bh, (short)0, c, false, false);
        c = __builtin_amdgcn_wmma_f32_16x16x32_f16(false, al, false, bh, (short)0, c, false, false);
        c = __builtin_amdgcn_wmma_f32_16x16x32_f16(false, ah, false, bl, (short)0, c, false, false);
        int ncol = ntile * 16 + lr;
        size_t rowbase = (size_t)rb * 16 + hs * 8;
        #pragma unroll
        for (int r = 0; r < 8; ++r) {
            out[(rowbase + r) * N_TIME + ncol] = c[r] * (1.0f / 4096.0f);
        }
    }
}

extern "C" void kernel_launch(void* const* d_in, const int* in_sizes, int n_in,
                              void* d_out, int out_size, void* d_ws, size_t ws_size,
                              hipStream_t stream) {
    const float* x  = (const float*)d_in[0];
    const float* wr = (const float*)d_in[1];
    const float* wi = (const float*)d_in[2];
    float* out = (float*)d_out;

    char* ws = (char*)d_ws;                       // 3 MB total
    unsigned* bas_hi  = (unsigned*)(ws + (size_t)0);
    unsigned* bas_lo  = (unsigned*)(ws + (size_t)512 * 1024);
    unsigned* binv_hi = (unsigned*)(ws + (size_t)1024 * 1024);
    unsigned* binv_lo = (unsigned*)(ws + (size_t)1536 * 1024);
    float*    Xw      = (float*)   (ws + (size_t)2048 * 1024);
    unsigned* c2h     = (unsigned*)(ws + (size_t)2560 * 1024);
    unsigned* c2l     = (unsigned*)(ws + (size_t)2816 * 1024);

    gen_bases<<<(BAS_DW + BINV_DW) / 256, 256, 0, stream>>>(bas_hi, bas_lo, binv_hi, binv_lo);
    phaseA<<<RB_COUNT, 256, 0, stream>>>(x, bas_hi, bas_lo, Xw);
    phaseB<<<BATCH, 1024, 0, stream>>>(Xw, wr, wi, c2h, c2l);
    phaseC<<<dim3(RB_COUNT, 16), 256, 0, stream>>>(c2h, c2l, binv_hi, binv_lo, out);
}
